// Transformer_38216618999881
// MI455X (gfx1250) — matrix operand order repaired and checked by reading, not compiled, not run on hardware
//
#include <hip/hip_runtime.h>

typedef __attribute__((ext_vector_type(16))) __bf16 v16bf;
typedef __attribute__((ext_vector_type(8)))  float  v8f;
typedef int v4i32 __attribute__((vector_size(16)));

namespace {
constexpr int kLayers = 4;
constexpr int kD      = 256;          // emb dim
constexpr int kHeads  = 8;
constexpr int kDH     = 256;          // per-head dim (module quirk: == emb dim)
constexpr int kInner  = 2048;         // kD * kHeads
constexpr int kQkvW   = 3 * kInner;   // 6144
constexpr int kMlp    = 1024;
constexpr int kB      = 8;
constexpr int kN      = 1024;
constexpr int kRows   = kB * kN;      // 8192
constexpr float kEps  = 1e-5f;
constexpr float kScale = 0.0625f;     // 256^-0.5
constexpr int FA_BQ = 64;             // query tile rows
constexpr int FA_BK = 32;             // key tile rows
constexpr int KPAD  = kDH + 8;        // 264: padded stride for Q/K tiles
constexpr int VPAD  = FA_BK + 8;      // 40: padded stride for transposed V / P
constexpr int SPAD  = FA_BK + 1;      // 33
}

// ---------------- async global->LDS (ASYNCcnt path), guarded ----------------
#if defined(__gfx1250__) &&                                                    \
    __has_builtin(__builtin_amdgcn_global_load_async_to_lds_b128) &&           \
    __has_builtin(__builtin_amdgcn_s_wait_asynccnt)
#define HAVE_ASYNC_LDS 1
#else
#define HAVE_ASYNC_LDS 0
#endif

#if HAVE_ASYNC_LDS
typedef __attribute__((address_space(1))) v4i32 as1_v4i32;
typedef __attribute__((address_space(3))) v4i32 as3_v4i32;
__device__ __forceinline__ void async_cp16(__bf16* lds_dst, const __bf16* g_src) {
  __builtin_amdgcn_global_load_async_to_lds_b128(
      (as1_v4i32*)(unsigned long long)g_src,
      (as3_v4i32*)(unsigned)(unsigned long long)lds_dst,
      0, 0);
}
#define ASYNC_WAIT() __builtin_amdgcn_s_wait_asynccnt(0)
#else
__device__ __forceinline__ void async_cp16(__bf16* lds_dst, const __bf16* g_src) {
  *(uint4*)lds_dst = *(const uint4*)g_src;
}
#define ASYNC_WAIT() ((void)0)
#endif

// ---------------- WMMA helpers ----------------
__device__ __forceinline__ v8f wmma_bf16(v16bf a, v16bf b, v8f c) {
  return __builtin_amdgcn_wmma_f32_16x16x32_bf16(false, a, false, b, (short)0, c,
                                                 false, false);
}

// A fragment: 16x32 bf16 tile (rows row0.., k k0..) from row-major storage.
// ISA layout: lane m=L&15; e<8 -> k=koff+e, e>=8 -> k=koff+16+(e-8), koff=8*(L>=16).
// => two contiguous 16B runs per lane.
__device__ __forceinline__ v16bf frag_a_vec(const __bf16* p, int row0, int k0,
                                            int ld, int lane) {
  const int m = lane & 15;
  const int koff = (lane & 16) ? 8 : 0;
  union { uint4 q[2]; v16bf v; } u;
  const __bf16* base = p + (row0 + m) * ld + k0 + koff;
  u.q[0] = *(const uint4*)base;
  u.q[1] = *(const uint4*)(base + 16);
  return u.v;
}

// B fragment: B[k][n] with K-MAJOR storage p[n][k] (stride ld).
// ISA layout: n=L&15; e<8 -> k=koff+e, e>=8 -> k=koff+16+(e-8), koff=8*(L>=16).
__device__ __forceinline__ v16bf frag_b_kmaj(const __bf16* p, int k0, int col0,
                                             int ld, int lane) {
  const int n = lane & 15;
  const int koff = (lane & 16) ? 8 : 0;
  union { uint4 q[2]; v16bf v; } u;
  const __bf16* base = p + (col0 + n) * ld + k0 + koff;
  u.q[0] = *(const uint4*)base;
  u.q[1] = *(const uint4*)(base + 16);
  return u.v;
}

// ---------------- fp32 -> bf16 transpose (weights -> K-major), per launch ----
// out[l][n][k] = (bf16) in[l][k][n]; 32x32 tiles, 256 threads (32x8).
__global__ void __launch_bounds__(256)
cvt_t_f32_bf16(const float* __restrict__ in, __bf16* __restrict__ out, int K, int N) {
  __shared__ __bf16 t[32][33];
  const int l = blockIdx.z;
  const int k0 = blockIdx.y * 32, n0 = blockIdx.x * 32;
  const int tx = threadIdx.x & 31, ty = threadIdx.x >> 5;
  const size_t base = (size_t)l * K * N;
#pragma unroll
  for (int i = 0; i < 32; i += 8)
    t[ty + i][tx] = (__bf16)in[base + (size_t)(k0 + ty + i) * N + n0 + tx];
  __syncthreads();
#pragma unroll
  for (int i = 0; i < 32; i += 8)
    out[base + (size_t)(n0 + ty + i) * K + k0 + tx] = t[tx][ty + i];
}

// ---------------- LayerNorm (fp32 in, bf16 out) ----------------
__global__ void __launch_bounds__(256)
layernorm_bf16(const float* __restrict__ x, const float* __restrict__ gw,
               const float* __restrict__ bw, __bf16* __restrict__ y) {
  const int row = blockIdx.x;
  const int t = threadIdx.x;  // 0..255 == kD
  __shared__ float red[256];
  const float v = x[(long)row * kD + t];
  red[t] = v;
  __syncthreads();
#pragma unroll
  for (int s = 128; s > 0; s >>= 1) {
    if (t < s) red[t] += red[t + s];
    __syncthreads();
  }
  const float mu = red[0] * (1.0f / kD);
  __syncthreads();
  const float d = v - mu;
  red[t] = d * d;
  __syncthreads();
#pragma unroll
  for (int s = 128; s > 0; s >>= 1) {
    if (t < s) red[t] += red[t + s];
    __syncthreads();
  }
  const float rstd = rsqrtf(red[0] * (1.0f / kD) + kEps);
  y[(long)row * kD + t] = (__bf16)(d * rstd * gw[t] + bw[t]);
}

// ---------------- GEMM: C[M,N] = A[M,K] @ Bt[N,K]^T (+bias)(+GELU)(+resid) ----
// Block tile 128x128x64; 8 waves, each 2x4 WMMA 16x16 tiles. B is K-major.
// Double-buffered: async-load tile t+1 into LDS while computing tile t.
template <bool HAS_BIAS, bool GELU, bool RESID, bool OUT_BF16>
__global__ void __launch_bounds__(256)
gemm_bf16_wmma(const __bf16* __restrict__ A, const __bf16* __restrict__ Bt,
               const float* __restrict__ bias, const float* __restrict__ resid,
               void* __restrict__ Cout, int M, int N, int K) {
  constexpr int BM = 128, BN = 128, BK = 64, LDP = BK + 8;  // 72
  __shared__ __align__(16) __bf16 sA[2][BM * LDP];
  __shared__ __align__(16) __bf16 sB[2][BN * LDP];  // K-major: [n][k]

  const int tid = threadIdx.x;
  const int lane = tid & 31;
  const int w = tid >> 5;
  const int wr = w & 3;   // 32-row group
  const int wc = w >> 2;  // 64-col group
  const int row0 = blockIdx.y * BM;
  const int col0 = blockIdx.x * BN;

  auto stage = [&](int buf, int k0) {
#pragma unroll
    for (int j = 0; j < 4; ++j) {
      const int v = tid * 4 + j;  // 1024 8-elem vectors each side
      const int ra = v >> 3, ca = (v & 7) << 3;
      async_cp16(&sA[buf][ra * LDP + ca], &A[(long)(row0 + ra) * K + k0 + ca]);
      async_cp16(&sB[buf][ra * LDP + ca], &Bt[(long)(col0 + ra) * K + k0 + ca]);
    }
  };

  v8f acc[2][4];
#pragma unroll
  for (int rt = 0; rt < 2; ++rt)
#pragma unroll
    for (int ct = 0; ct < 4; ++ct)
#pragma unroll
      for (int i = 0; i < 8; ++i) acc[rt][ct][i] = 0.0f;

  stage(0, 0);
  ASYNC_WAIT();
  __syncthreads();

  const int nk = K / BK;
  for (int t = 0; t < nk; ++t) {
    const int cur = t & 1;
    if (t + 1 < nk) stage(cur ^ 1, (t + 1) * BK);  // overlaps compute below
#pragma unroll
    for (int kk = 0; kk < BK; kk += 32) {
      const v16bf a0 = frag_a_vec(sA[cur], wr * 32, kk, LDP, lane);
      const v16bf a1 = frag_a_vec(sA[cur], wr * 32 + 16, kk, LDP, lane);
#pragma unroll
      for (int ct = 0; ct < 4; ++ct) {
        const v16bf bb = frag_b_kmaj(sB[cur], kk, wc * 64 + ct * 16, LDP, lane);
        acc[0][ct] = wmma_bf16(a0, bb, acc[0][ct]);
        acc[1][ct] = wmma_bf16(a1, bb, acc[1][ct]);
      }
    }
    ASYNC_WAIT();
    __syncthreads();
  }

  const int roff = (lane & 16) ? 8 : 0;
  const int cl = lane & 15;
#pragma unroll
  for (int rt = 0; rt < 2; ++rt)
#pragma unroll
    for (int ct = 0; ct < 4; ++ct)
#pragma unroll
      for (int i = 0; i < 8; ++i) {
        const int r = row0 + wr * 32 + rt * 16 + i + roff;
        const int c = col0 + wc * 64 + ct * 16 + cl;
        float v = acc[rt][ct][i];
        if constexpr (HAS_BIAS) v += bias[c];
        if constexpr (GELU) v = 0.5f * v * (1.0f + erff(v * 0.70710678118654752f));
        if constexpr (RESID) v += resid[(long)r * N + c];
        if constexpr (OUT_BF16)
          ((__bf16*)Cout)[(long)r * N + c] = (__bf16)v;
        else
          ((float*)Cout)[(long)r * N + c] = v;
      }
}

// ---------------- Flash attention ----------------
// One block per (query-tile, b, h). Q fragments in registers, K tile async-to-LDS
// (row-major = K-major for Q.K^T), V tile transposed at staging (K-major for
// P.V) with next-tile V prefetched into registers, online softmax in LDS.
__global__ void __launch_bounds__(256)
flash_attn_bf16(const __bf16* __restrict__ qkv, __bf16* __restrict__ outp) {
  const int qt = blockIdx.x;       // 0..15
  const int b = blockIdx.y >> 3;   // 0..7
  const int h = blockIdx.y & 7;    // 0..7
  const int tid = threadIdx.x;
  const int lane = tid & 31;
  const int w = tid >> 5;
  const int wr = w & 3;   // 16-row group of the 64-row query tile
  const int wc = w >> 2;  // 0/1: S col half (16) / O col half (128)

  // sK: 32 x KPAD ; sVt: 256 x VPAD (transposed V). Q staging aliases both.
  __shared__ __align__(16) __bf16 smemKV[FA_BK * KPAD + kDH * VPAD];
  __shared__ __align__(16) float sS[FA_BQ * SPAD];
  __shared__ __align__(16) __bf16 sP[FA_BQ * VPAD];
  __shared__ float rowM[FA_BQ], rowL[FA_BQ], rowC[FA_BQ];
  __bf16* sK = smemKV;
  __bf16* sVt = smemKV + FA_BK * KPAD;
  __bf16* sQstage = smemKV;  // consumed before sK/sVt first written

  const __bf16* Qg = qkv + (long)(b * kN) * kQkvW + h * kDH;
  const __bf16* Kg = Qg + kInner;
  const __bf16* Vg = Qg + 2 * kInner;

  // Stage the 64x256 Q tile, pull per-wave A fragments into registers.
#pragma unroll
  for (int j = 0; j < 8; ++j) {
    const int v = tid * 8 + j;  // 2048 8-elem vectors
    const int r = v >> 5, c8 = (v & 31) << 3;
    async_cp16(&sQstage[r * KPAD + c8],
               &Qg[(long)(qt * FA_BQ + r) * kQkvW + c8]);
  }
  if (tid < FA_BQ) { rowM[tid] = -1.0e30f; rowL[tid] = 0.0f; }
  ASYNC_WAIT();
  __syncthreads();

  v16bf qf[kDH / 32];
#pragma unroll
  for (int d0 = 0; d0 < kDH; d0 += 32)
    qf[d0 / 32] = frag_a_vec(sQstage, wr * 16, d0, KPAD, lane);

  v8f o[8];
#pragma unroll
  for (int t = 0; t < 8; ++t)
#pragma unroll
    for (int i = 0; i < 8; ++i) o[t][i] = 0.0f;

  const int roff = (lane & 16) ? 8 : 0;
  const int cl = lane & 15;

  // V software pipeline: vreg holds tile kt's V vectors during iteration kt.
  uint4 vreg[4];
  auto loadV = [&](int kt) {
#pragma unroll
    for (int j = 0; j < 4; ++j) {
      const int v = tid * 4 + j;
      const int r = v >> 5, c8 = (v & 31) << 3;
      vreg[j] = *(const uint4*)&Vg[(long)(kt * FA_BK + r) * kQkvW + c8];
    }
  };
  loadV(0);

  const int nkt = kN / FA_BK;
  for (int kt = 0; kt < nkt; ++kt) {
    __syncthreads();  // Q frags extracted / previous P@V finished
#pragma unroll
    for (int j = 0; j < 4; ++j) {
      const int v = tid * 4 + j;  // 1024 vectors per 32x256 tile
      const int r = v >> 5, c8 = (v & 31) << 3;
      async_cp16(&sK[r * KPAD + c8],
                 &Kg[(long)(kt * FA_BK + r) * kQkvW + c8]);
      union { uint4 q; __bf16 e[8]; } u;  // V transposed into sVt[d][key]
      u.q = vreg[j];
#pragma unroll
      for (int jj = 0; jj < 8; ++jj) sVt[(c8 + jj) * VPAD + r] = u.e[jj];
    }
    if (kt + 1 < nkt) loadV(kt + 1);  // global loads overlap compute below
    ASYNC_WAIT();
    __syncthreads();

    // S tile (16x16 per wave): rows wr*16.., keys wc*16..; contract over 256.
    v8f s;
#pragma unroll
    for (int i = 0; i < 8; ++i) s[i] = 0.0f;
#pragma unroll
    for (int d0 = 0; d0 < kDH; d0 += 32) {
      const v16bf bk = frag_b_kmaj(sK, d0, wc * 16, KPAD, lane);
      s = wmma_bf16(qf[d0 / 32], bk, s);
    }
#pragma unroll
    for (int i = 0; i < 8; ++i)
      sS[(wr * 16 + i + roff) * SPAD + wc * 16 + cl] = s[i] * kScale;
    __syncthreads();

    // Online softmax, one thread per query row.
    if (tid < FA_BQ) {
      const float mprev = rowM[tid];
      float mnew = mprev;
#pragma unroll 8
      for (int c = 0; c < FA_BK; ++c) mnew = fmaxf(mnew, sS[tid * SPAD + c]);
      const float corr = __expf(mprev - mnew);
      float sum = 0.0f;
#pragma unroll 8
      for (int c = 0; c < FA_BK; ++c) {
        const float p = __expf(sS[tid * SPAD + c] - mnew);
        sP[tid * VPAD + c] = (__bf16)p;
        sum += p;
      }
      rowL[tid] = rowL[tid] * corr + sum;
      rowM[tid] = mnew;
      rowC[tid] = corr;
    }
    __syncthreads();

    // Rescale O accumulators and add P@V (one K=32 WMMA step).
    float cr[8];
#pragma unroll
    for (int i = 0; i < 8; ++i) cr[i] = rowC[wr * 16 + i + roff];
    const v16bf pa = frag_a_vec(sP, wr * 16, 0, VPAD, lane);
#pragma unroll
    for (int t = 0; t < 8; ++t) {
#pragma unroll
      for (int i = 0; i < 8; ++i) o[t][i] *= cr[i];
      const v16bf bv = frag_b_kmaj(sVt, 0, wc * 128 + t * 16, VPAD, lane);
      o[t] = wmma_bf16(pa, bv, o[t]);
    }
  }

  float inv[8];
#pragma unroll
  for (int i = 0; i < 8; ++i) inv[i] = 1.0f / rowL[wr * 16 + i + roff];
#pragma unroll
  for (int t = 0; t < 8; ++t)
#pragma unroll
    for (int i = 0; i < 8; ++i) {
      const int row = qt * FA_BQ + wr * 16 + i + roff;
      const int col = wc * 128 + t * 16 + cl;
      outp[(long)(b * kN + row) * kInner + h * kDH + col] =
          (__bf16)(o[t][i] * inv[i]);
    }
}

// ---------------- host-side orchestration ----------------
extern "C" void kernel_launch(void* const* d_in, const int* in_sizes, int n_in,
                              void* d_out, int out_size, void* d_ws, size_t ws_size,
                              hipStream_t stream) {
  (void)in_sizes; (void)n_in; (void)out_size; (void)ws_size;
  const float* inputs = (const float*)d_in[0];
  const float* ln1_g  = (const float*)d_in[1];
  const float* ln1_b  = (const float*)d_in[2];
  const float* w_qkv  = (const float*)d_in[3];
  const float* w_proj = (const float*)d_in[4];
  const float* b_proj = (const float*)d_in[5];
  const float* ln2_g  = (const float*)d_in[6];
  const float* ln2_b  = (const float*)d_in[7];
  const float* w1     = (const float*)d_in[8];
  const float* b1     = (const float*)d_in[9];
  const float* w2     = (const float*)d_in[10];
  const float* b2     = (const float*)d_in[11];

  char* ws = (char*)d_ws;
  size_t off = 0;
  auto carve = [&](size_t bytes) -> void* {
    void* p = ws + off;
    off += (bytes + 255) & ~(size_t)255;
    return p;
  };

  float*  xbuf     = (float*)carve((size_t)kRows * kD * 4);
  __bf16* ln_out   = (__bf16*)carve((size_t)kRows * kD * 2);
  __bf16* qkv      = (__bf16*)carve((size_t)kRows * kQkvW * 2);
  __bf16* attnout  = (__bf16*)carve((size_t)kRows * kInner * 2);
  __bf16* h1       = (__bf16*)carve((size_t)kRows * kMlp * 2);
  __bf16* wqkv_t   = (__bf16*)carve((size_t)kLayers * kD * kQkvW * 2);   // [N][K]
  __bf16* wproj_t  = (__bf16*)carve((size_t)kLayers * kInner * kD * 2);  // [N][K]
  __bf16* w1_t     = (__bf16*)carve((size_t)kLayers * kD * kMlp * 2);    // [N][K]
  __bf16* w2_t     = (__bf16*)carve((size_t)kLayers * kMlp * kD * 2);    // [N][K]

  // weights: fp32 [l][K][N] -> bf16 K-major [l][N][K]
  cvt_t_f32_bf16<<<dim3(kQkvW / 32, kD / 32, kLayers),   256, 0, stream>>>(w_qkv,  wqkv_t,  kD,    kQkvW);
  cvt_t_f32_bf16<<<dim3(kD / 32, kInner / 32, kLayers),  256, 0, stream>>>(w_proj, wproj_t, kInner, kD);
  cvt_t_f32_bf16<<<dim3(kMlp / 32, kD / 32, kLayers),    256, 0, stream>>>(w1,     w1_t,    kD,    kMlp);
  cvt_t_f32_bf16<<<dim3(kD / 32, kMlp / 32, kLayers),    256, 0, stream>>>(w2,     w2_t,    kMlp,  kD);

  (void)hipMemcpyAsync(xbuf, inputs, (size_t)kRows * kD * 4,
                       hipMemcpyDeviceToDevice, stream);

  for (int l = 0; l < kLayers; ++l) {
    layernorm_bf16<<<kRows, 256, 0, stream>>>(xbuf, ln1_g + l * kD, ln1_b + l * kD,
                                              ln_out);
    gemm_bf16_wmma<false, false, false, true>
        <<<dim3(kQkvW / 128, kRows / 128), 256, 0, stream>>>(
            ln_out, wqkv_t + (size_t)l * kD * kQkvW, nullptr, nullptr, qkv,
            kRows, kQkvW, kD);
    flash_attn_bf16<<<dim3(kN / FA_BQ, kB * kHeads), 256, 0, stream>>>(qkv, attnout);
    gemm_bf16_wmma<true, false, true, false>
        <<<dim3(kD / 128, kRows / 128), 256, 0, stream>>>(
            attnout, wproj_t + (size_t)l * kInner * kD, b_proj + l * kD, xbuf, xbuf,
            kRows, kD, kInner);
    layernorm_bf16<<<kRows, 256, 0, stream>>>(xbuf, ln2_g + l * kD, ln2_b + l * kD,
                                              ln_out);
    gemm_bf16_wmma<true, true, false, true>
        <<<dim3(kMlp / 128, kRows / 128), 256, 0, stream>>>(
            ln_out, w1_t + (size_t)l * kD * kMlp, b1 + l * kMlp, nullptr, h1,
            kRows, kMlp, kD);
    gemm_bf16_wmma<true, false, true, false>
        <<<dim3(kD / 128, kRows / 128), 256, 0, stream>>>(
            h1, w2_t + (size_t)l * kMlp * kD, b2 + l * kD, xbuf, xbuf,
            kRows, kD, kMlp);
  }

  (void)hipMemcpyAsync(d_out, xbuf, (size_t)kRows * kD * 4,
                       hipMemcpyDeviceToDevice, stream);
}
